// LSTMClassifier_3118146257465
// MI455X (gfx1250) — matrix operand order repaired and checked by reading, not compiled, not run on hardware
//
#include <hip/hip_runtime.h>
#include <hip/hip_bf16.h>

// ---------------------------------------------------------------------------
// Fused LSTM (B=1024,T=512,F=128,H=32) + MLP head for MI455X (gfx1250).
// - x (256MB fp32) streamed once with nontemporal loads, software-pipelined
//   into LDS as bf16 (prefetch regs hide global latency under WMMA phase).
// - gates = bias + x@w_ih^T + h@w_hh^T via v_wmma_f32_16x16x32_bf16
//   (4 K=32 chunks for F=128, 1 chunk for H=32), fp32 accumulation.
// - weight fragments + bias vector pinned in registers via empty asm so the
//   backend cannot rematerialize them inside the recurrent loop.
// - cell state c lives in registers; h shared via LDS (bf16 for WMMA A).
// - activations use native TRANS32 ops (v_exp_f32 / v_rcp_f32 / v_tanh_f32).
// Grid: 64 blocks x 256 threads (8 wave32); wave w owns gate columns
// [16w,16w+16) of the [16x128] gate tile for its 16 batch rows.
// ---------------------------------------------------------------------------

typedef __attribute__((ext_vector_type(16))) __bf16 v16bf;
typedef __attribute__((ext_vector_type(2)))  __bf16 v2bf;
typedef __attribute__((ext_vector_type(8)))  float  v8f;
typedef __attribute__((ext_vector_type(4)))  float  vf4;
typedef __attribute__((ext_vector_type(4)))  unsigned int vu4;

union Frag {
  v16bf          v;
  unsigned short u[16];
  unsigned int   d[8];
  vu4            q[2];
};

// Pack two f32 into one dword of bf16 (lo in [15:0], hi in [31:16]).
__device__ __forceinline__ unsigned int pack2_bf16(float lo, float hi) {
#if __has_builtin(__builtin_amdgcn_cvt_pk_bf16_f32)
  v2bf r = __builtin_amdgcn_cvt_pk_bf16_f32(lo, hi);
  return __builtin_bit_cast(unsigned int, r);
#else
  unsigned int a = __float_as_uint(lo);
  unsigned int b = __float_as_uint(hi);
  a += 0x7FFFu + ((a >> 16) & 1u);    // RNE
  b += 0x7FFFu + ((b >> 16) & 1u);
  return (a >> 16) | (b & 0xFFFF0000u);
#endif
}

__device__ __forceinline__ unsigned short bf16_bits(float f) {
  return (unsigned short)pack2_bf16(f, f);
}

__device__ __forceinline__ float fast_rcp(float x) {
#if __has_builtin(__builtin_amdgcn_rcpf)
  return __builtin_amdgcn_rcpf(x);
#else
  return 1.0f / x;
#endif
}

__device__ __forceinline__ float fast_sigmoid(float x) {
  // rcp(1+exp(-x)); exp->inf gives rcp(inf)=0, correct saturation.
  return fast_rcp(1.0f + __expf(-x));
}

__device__ __forceinline__ float fast_tanh(float x) {
#if __has_builtin(__builtin_amdgcn_tanhf)
  return __builtin_amdgcn_tanhf(x);
#elif __has_builtin(__builtin_amdgcn_tanh_f32)
  return __builtin_amdgcn_tanh_f32(x);
#else
  float xc = fminf(fmaxf(x, -15.0f), 15.0f);
  float e  = __expf(-2.0f * xc);
  return (1.0f - e) * fast_rcp(1.0f + e);
#endif
}

constexpr int Bsz  = 1024;
constexpr int Tlen = 512;
constexpr int Fdim = 128;
constexpr int Hdim = 32;
constexpr int Gdim = 4 * Hdim;   // 128 gate rows
constexpr int TILE_B = 16;

__global__ __launch_bounds__(256) void lstm_head_fused(
    const float* __restrict__ x,      // [B, T, F]
    const float* __restrict__ w_ih,   // [4H, F]
    const float* __restrict__ w_hh,   // [4H, H]
    const float* __restrict__ b_ih,   // [4H]
    const float* __restrict__ b_hh,   // [4H]
    const float* __restrict__ w1,     // [16, H]
    const float* __restrict__ b1,     // [16]
    const float* __restrict__ w2,     // [1, 16]
    const float* __restrict__ b2,     // [1]
    float* __restrict__ out)          // [B]
{
  __shared__ unsigned short x_lds[TILE_B][Fdim];   // bf16 x tile          (4KB)
  __shared__ float gates_lds[TILE_B][Gdim];        // raw fp32 gates       (8KB)
  __shared__ unsigned short h_bf[TILE_B][Hdim];    // bf16 hidden (WMMA A) (1KB)
  __shared__ float h_f32[TILE_B][Hdim];            // fp32 hidden (head)   (2KB)

  const int tid  = threadIdx.x;
  const int wave = tid >> 5;      // 0..7 -> gate column tile
  const int lane = tid & 31;
  const int sub  = lane >> 4;     // half-wave select
  const int nn   = lane & 15;     // N column / A row within tile
  const int b0   = blockIdx.x * TILE_B;

  // ---- loop-invariant B fragments (weights), packed dwords in registers -
  // B(32x16) layout: lane -> N = nn, K = sub*16 + e (contiguous 16 floats)
  const int g = wave * 16 + nn;                       // gate row 0..127
  Frag Bx[4];                                         // w_ih^T, 4 K-chunks
  {
    const float* wrow = w_ih + g * Fdim;
    #pragma unroll
    for (int kk = 0; kk < 4; ++kk) {
      const int kbase = kk * 32 + sub * 16;
      #pragma unroll
      for (int e = 0; e < 8; ++e)
        Bx[kk].d[e] = pack2_bf16(wrow[kbase + 2 * e], wrow[kbase + 2 * e + 1]);
    }
  }
  Frag Bh;                                            // w_hh^T, K = 32
  {
    const float* wrow = w_hh + g * Hdim + sub * 16;
    #pragma unroll
    for (int e = 0; e < 8; ++e)
      Bh.d[e] = pack2_bf16(wrow[2 * e], wrow[2 * e + 1]);
  }
  const float bias = b_ih[g] + b_hh[g];
  v8f bias_vec;
  #pragma unroll
  for (int v = 0; v < 8; ++v) bias_vec[v] = bias;

  // Pin packed fragments + bias splat: opaque to the optimizer so they stay
  // live as packed registers and are never rematerialized inside the loop.
  asm volatile(""
      : "+v"(Bx[0].q[0]), "+v"(Bx[0].q[1]),
        "+v"(Bx[1].q[0]), "+v"(Bx[1].q[1]),
        "+v"(Bx[2].q[0]), "+v"(Bx[2].q[1]),
        "+v"(Bx[3].q[0]), "+v"(Bx[3].q[1]),
        "+v"(Bh.q[0]),    "+v"(Bh.q[1]),
        "+v"(bias_vec));

  // ---- x prefetch pipeline: this thread always stages the same 2 vf4 ----
  const int elemA = tid * 8;                  // 0..2047 over 16x128 tile
  const int rA    = elemA >> 7;
  const int colA  = elemA & (Fdim - 1);
  const float* xp = x + (size_t)(b0 + rA) * Tlen * Fdim + colA; // +t*Fdim/step
  vf4 pf0 = __builtin_nontemporal_load((const vf4*)xp);
  vf4 pf1 = __builtin_nontemporal_load((const vf4*)(xp + 4));
  xp += Fdim;

  // ---- recurrent state: c in registers, h in LDS ------------------------
  const int rE = tid >> 5;        // elementwise rows: rE and rE+8
  const int jE = tid & (Hdim - 1);
  float c0 = 0.0f, c1 = 0.0f;
  for (int idx = tid; idx < TILE_B * Hdim; idx += 256) {
    h_bf[idx >> 5][idx & (Hdim - 1)] = 0;
  }

  // ---- time loop: iteration t stages x[t] + applies gates of t-1 --------
  for (int t = 0; t <= Tlen; ++t) {
    if (t < Tlen) {
      // stage prefetched x[t] as bf16 (16 bytes per thread)
      unsigned int* dst = (unsigned int*)&x_lds[rA][colA];
      dst[0] = pack2_bf16(pf0[0], pf0[1]);
      dst[1] = pack2_bf16(pf0[2], pf0[3]);
      dst[2] = pack2_bf16(pf1[0], pf1[1]);
      dst[3] = pack2_bf16(pf1[2], pf1[3]);
      if (t + 1 < Tlen) {   // issue next loads; complete under WMMA phase
        pf0 = __builtin_nontemporal_load((const vf4*)xp);
        pf1 = __builtin_nontemporal_load((const vf4*)(xp + 4));
        xp += Fdim;
      }
    }
    if (t > 0) {
      // elementwise LSTM cell update from gates of step t-1 (2 elems/thread)
      #pragma unroll
      for (int i = 0; i < 2; ++i) {
        const int r = rE + i * 8;
        const float ig = gates_lds[r][jE];
        const float fg = gates_lds[r][Hdim + jE];
        const float gg = gates_lds[r][2 * Hdim + jE];
        const float og = gates_lds[r][3 * Hdim + jE];
        const float i_ = fast_sigmoid(ig);
        const float f_ = fast_sigmoid(fg);
        const float g_ = fast_tanh(gg);
        const float o_ = fast_sigmoid(og);
        float& c = (i == 0) ? c0 : c1;
        c = f_ * c + i_ * g_;
        const float h = o_ * fast_tanh(c);
        h_bf[r][jE] = bf16_bits(h);
        if (t == Tlen) h_f32[r][jE] = h;   // only the final hidden state
      }
    }
    __syncthreads();

    // gates(t) = bias + x_tile @ w_ih^T + h @ w_hh^T  via WMMA
    if (t < Tlen) {
      // A(16x32 bf16): lane (m=nn, sub); elems 0..15 @ K=kk*32+sub*16+e
      // (contiguous, matches B) -> two 16B LDS loads per chunk.
      Frag A0;
      A0.q[0] = *(const vu4*)&x_lds[nn][sub * 16];
      A0.q[1] = *(const vu4*)&x_lds[nn][sub * 16 + 8];
      // first WMMA reads the persistent bias vector directly as C
      v8f acc = __builtin_amdgcn_wmma_f32_16x16x32_bf16(
          false, A0.v, false, Bx[0].v, (short)0, bias_vec, false, false);
      #pragma unroll
      for (int kk = 1; kk < 4; ++kk) {
        Frag A;
        A.q[0] = *(const vu4*)&x_lds[nn][kk * 32 + sub * 16];
        A.q[1] = *(const vu4*)&x_lds[nn][kk * 32 + sub * 16 + 8];
        acc = __builtin_amdgcn_wmma_f32_16x16x32_bf16(
            false, A.v, false, Bx[kk].v, (short)0, acc, false, false);
      }
      {
        Frag Ah;
        Ah.q[0] = *(const vu4*)&h_bf[nn][sub * 16];
        Ah.q[1] = *(const vu4*)&h_bf[nn][sub * 16 + 8];
        acc = __builtin_amdgcn_wmma_f32_16x16x32_bf16(
            false, Ah.v, false, Bh.v, (short)0, acc, false, false);
      }

      // C/D layout: lane -> col N=nn, rows M = v + 8*sub
      #pragma unroll
      for (int v = 0; v < 8; ++v)
        gates_lds[v + 8 * sub][wave * 16 + nn] = acc[v];
    }
    __syncthreads();
  }

  // ---- head: Linear(H,16) -> ReLU -> Linear(16,1) ----------------------
  if (tid < TILE_B) {
    const int r = tid;
    float logit = b2[0];
    #pragma unroll
    for (int k = 0; k < 16; ++k) {
      float s = b1[k];
      #pragma unroll
      for (int j = 0; j < Hdim; ++j)
        s += w1[k * Hdim + j] * h_f32[r][j];
      s = fmaxf(s, 0.0f);
      logit += w2[k] * s;
    }
    out[b0 + r] = logit;
  }
}

extern "C" void kernel_launch(void* const* d_in, const int* in_sizes, int n_in,
                              void* d_out, int out_size, void* d_ws, size_t ws_size,
                              hipStream_t stream) {
  (void)in_sizes; (void)n_in; (void)out_size; (void)d_ws; (void)ws_size;
  const float* x    = (const float*)d_in[0];
  const float* w_ih = (const float*)d_in[1];
  const float* w_hh = (const float*)d_in[2];
  const float* b_ih = (const float*)d_in[3];
  const float* b_hh = (const float*)d_in[4];
  const float* w1   = (const float*)d_in[5];
  const float* b1   = (const float*)d_in[6];
  const float* w2   = (const float*)d_in[7];
  const float* b2   = (const float*)d_in[8];
  float* out = (float*)d_out;

  dim3 grid(Bsz / TILE_B);   // 64 workgroups
  dim3 block(256);           // 8 wave32
  lstm_head_fused<<<grid, block, 0, stream>>>(x, w_ih, w_hh, b_ih, b_hh,
                                              w1, b1, w2, b2, out);
}
